// QuadraticSplineLayer_72181220376722
// MI455X (gfx1250) — compile-verified, matrix operand order checked by reading, not run on hardware
//
#include <hip/hip_runtime.h>

// Problem constants
#define NB   8192
#define NS   512
#define NK   8
#define NH   512
#define NCOL 8704           // S*(2K+1)
#define EPSV 1e-6f
#define BSTR 40             // LDS column stride in halves (80B) -> conflict-free frags

#define USE_ASYNC 1         // global_load_async_to_lds_b128 staging (CDNA5)

typedef __attribute__((ext_vector_type(16))) _Float16 v16h;
typedef __attribute__((ext_vector_type(8)))  _Float16 v8h;
typedef __attribute__((ext_vector_type(8)))  float    v8f;

// Workspace layout (bytes)
#define WS_XAH  0UL                     // 8192*512 f16  = 8,388,608
#define WS_HID  8388608UL               // 8192*512 f16  = 8,388,608
#define WS_W1T  16777216UL              // 512*512 f16   =   524,288
#define WS_W2T  17301504UL              // 8704*512 f16  = 8,912,896
#define WS_PART 26214400UL              // 8192*32 f32   = 1,048,576

static __device__ __forceinline__ v16h make_frag(const _Float16* p) {
  v8h lo = *(const v8h*)(p);
  v8h hi = *(const v8h*)(p + 16);
  return __builtin_shufflevector(lo, hi, 0,1,2,3,4,5,6,7,8,9,10,11,12,13,14,15);
}

#if USE_ASYNC
// Async memory->LDS copy, 16B per lane. LDS address = low 32 bits of the
// generic shared pointer (ISA 10.2).
static __device__ __forceinline__ void async_b128(const _Float16* g, _Float16* l) {
  unsigned lds = (unsigned)(unsigned long long)(void*)l;
  asm volatile("global_load_async_to_lds_b128 %0, %1, off"
               : : "v"(lds), "v"(g) : "memory");
}
static __device__ __forceinline__ void wait_async_zero() {
  asm volatile("s_wait_asynccnt 0x0" ::: "memory");
}
#endif

// WMMA chain over NT 16-col tiles with 2-deep rotating B-fragment buffers.
template <int NT>
static __device__ __forceinline__ void mma_chain(const _Float16* __restrict__ lb,
                                                 const v16h& a, v8f* acc) {
  v16h b0 = make_frag(lb);
  v16h b1 = make_frag(lb + 16 * BSTR);
#pragma unroll
  for (int t = 0; t < NT; ++t) {
    const int tn = (t + 2 < NT) ? (t + 2) : (NT - 1);   // clamp (2 redundant)
    v16h bn = make_frag(lb + tn * (16 * BSTR));
    acc[t] = __builtin_amdgcn_wmma_f32_16x16x32_f16(
        false, a, false, b0, (short)0, acc[t], false, false);
    b0 = b1; b1 = bn;
  }
}

// ---------------- prep: x_a - 0.5 -> f16, copy x_a to phi_out[:, :S] ---------
__global__ void prep_x_kernel(const float* __restrict__ x_input,
                              _Float16* __restrict__ xah,
                              float* __restrict__ phi_out) {
  int i = blockIdx.x * 256 + threadIdx.x;        // i < 8192*512 exactly
  int row = i >> 9, col = i & 511;
  float v = x_input[row * (2*NS) + col];
  xah[i] = (_Float16)(v - 0.5f);
  phi_out[row * (2*NS) + col] = v;
}

// ---------------- prep: transpose W (K=512 x N) -> WT (N x 512) f16 ----------
__global__ void prep_wt_kernel(const float* __restrict__ W,
                               _Float16* __restrict__ WT, int N) {
  int i = blockIdx.x * 256 + threadIdx.x;        // grid sized exactly N*512
  int n = i >> 9, k = i & 511;
  WT[i] = (_Float16)W[(long)k * N + n];
}

// ======================= B-tile staging (async -> LDS) =======================
// GEMM1: 128 cols * 4 chunks = 512 chunks (2/thread).
// GEMM2: 272 cols * 4 chunks = 1088 chunks (4/thread + 64-thread partial).
#if USE_ASYNC
static __device__ __forceinline__ void b1_stage(const _Float16* __restrict__ bsrc,
                                                int k, int tid, _Float16* dst) {
#pragma unroll
  for (int it = 0; it < 2; ++it) {
    int idx = tid + it * 256, col = idx >> 2, q = idx & 3;
    async_b128(bsrc + col * NS + k + q * 8, dst + col * BSTR + q * 8);
  }
}
static __device__ __forceinline__ void b2_stage(const _Float16* __restrict__ bsrc,
                                                int k, int tid, _Float16* dst) {
#pragma unroll
  for (int it = 0; it < 4; ++it) {
    int idx = tid + it * 256, col = idx >> 2, q = idx & 3;
    async_b128(bsrc + col * NH + k + q * 8, dst + col * BSTR + q * 8);
  }
  if (tid < 64) {
    int col = (tid + 1024) >> 2, q = tid & 3;
    async_b128(bsrc + col * NH + k + q * 8, dst + col * BSTR + q * 8);
  }
}
#else
static __device__ __forceinline__ void b1_stage(const _Float16* __restrict__ bsrc,
                                                int k, int tid, _Float16* dst) {
  v8h sr[2];
#pragma unroll
  for (int it = 0; it < 2; ++it) {
    int idx = tid + it * 256, col = idx >> 2, q = idx & 3;
    sr[it] = *(const v8h*)(bsrc + col * NS + k + q * 8);
  }
#pragma unroll
  for (int it = 0; it < 2; ++it) {
    int idx = tid + it * 256, col = idx >> 2, q = idx & 3;
    *(v8h*)(dst + col * BSTR + q * 8) = sr[it];
  }
}
static __device__ __forceinline__ void b2_stage(const _Float16* __restrict__ bsrc,
                                                int k, int tid, _Float16* dst) {
  v8h sr[5];
#pragma unroll
  for (int it = 0; it < 4; ++it) {
    int idx = tid + it * 256, col = idx >> 2, q = idx & 3;
    sr[it] = *(const v8h*)(bsrc + col * NH + k + q * 8);
  }
  if (tid < 64) {
    int col = (tid + 1024) >> 2, q = tid & 3;
    sr[4] = *(const v8h*)(bsrc + col * NH + k + q * 8);
  }
#pragma unroll
  for (int it = 0; it < 4; ++it) {
    int idx = tid + it * 256, col = idx >> 2, q = idx & 3;
    *(v8h*)(dst + col * BSTR + q * 8) = sr[it];
  }
  if (tid < 64) {
    int col = (tid + 1024) >> 2, q = tid & 3;
    *(v8h*)(dst + col * BSTR + q * 8) = sr[4];
  }
}
static __device__ __forceinline__ void wait_async_zero() {}
#endif

// ---------------- GEMM1: hid = tanh((x-0.5) @ W1 + b1), f16 out -------------
__global__ __launch_bounds__(256)
void gemm1_tanh_kernel(const _Float16* __restrict__ Xah,
                       const _Float16* __restrict__ W1T,
                       const float* __restrict__ b1,
                       _Float16* __restrict__ hid) {
  __shared__ _Float16 bbuf[2][128 * BSTR];       // 2 x 10,240 B

  const int tid  = threadIdx.x;
  const int wave = tid >> 5;
  const int lane = tid & 31;
  const int m    = lane & 15;
  const int hw   = lane >> 4;
  const int koff = hw << 3;                      // 0 or 8
  const int rowBase = blockIdx.x * 128 + wave * 16;
  const int colBase = blockIdx.y * 128;

  const _Float16* ap   = Xah + (size_t)(rowBase + m) * NS + koff;
  const _Float16* bsrc = W1T + (size_t)colBase * NS;

  v8f acc[8];
#pragma unroll
  for (int t = 0; t < 8; ++t) acc[t] = (v8f){0,0,0,0,0,0,0,0};

  b1_stage(bsrc, 0, tid, bbuf[0]);               // prologue: stage k=0 tile
  wait_async_zero();
  __syncthreads();

  v16h a = make_frag(ap);
  const int lboff = m * BSTR + koff;
#pragma unroll
  for (int kidx = 0; kidx < 15; ++kidx) {        // fully unrolled steady loop
    const int k = kidx * 32;
    const int cur = kidx & 1;
    b1_stage(bsrc, k + 32, tid, bbuf[cur ^ 1]);  // async: issue only
    v16h an = make_frag(ap + k + 32);            // next A frag
    __builtin_amdgcn_sched_barrier(0);           // pin issue before the chain

    mma_chain<8>(&bbuf[cur][lboff], a, acc);
    a = an;
    wait_async_zero();
    __syncthreads();
  }
  mma_chain<8>(&bbuf[1][lboff], a, acc);         // peeled final k-step (buf 1)

  const int r0 = rowBase + (hw << 3);
#pragma unroll
  for (int t = 0; t < 8; ++t) {
    int n = colBase + t * 16 + m;
    float bias = b1[n];
#pragma unroll
    for (int r = 0; r < 8; ++r)
      hid[(size_t)(r0 + r) * NH + n] = (_Float16)tanhf(acc[t][r] + bias);
  }
}

// -------- GEMM2 fused: net=tanh(hid@W2+b2) tile in LDS -> spline math --------
// Block: 256 thr (8 waves). Tile 64 rows x 272 cols (= 16 spline channels).
// wave = (row-tile rt = wave>>1, col-group cg = wave&1); cg0 -> tiles 0..8,
// cg1 -> tiles 8..16 (tile 8 duplicated, bitwise-identical -> benign).
__global__ __launch_bounds__(256)
void gemm2_spline_kernel(const _Float16* __restrict__ hid,
                         const _Float16* __restrict__ W2T,
                         const float* __restrict__ b2,
                         const float* __restrict__ x_input,
                         float* __restrict__ out,        // phi_out base
                         float* __restrict__ part) {
  union SMem {
    _Float16 bbuf[2][272 * BSTR];                // 2 x 21,760 B (k-loop)
    struct { float net_s[32 * 272]; float red[256]; } ep;  // epilogue
  };
  __shared__ SMem smem;

  const int tid  = threadIdx.x;
  const int wave = tid >> 5;
  const int lane = tid & 31;
  const int m    = lane & 15;
  const int hw   = lane >> 4;
  const int koff = hw << 3;
  const int rt   = wave >> 1;                    // row tile 0..3
  const int cg   = wave & 1;                     // col group 0/1
  const int rowBase = blockIdx.x * 64;
  const int colBase = blockIdx.y * 272;          // = blockIdx.y*16 spline ch.
  const int waveRow = rowBase + rt * 16;

  const _Float16* ap   = hid + (size_t)(waveRow + m) * NH + koff;
  const _Float16* bsrc = W2T + (size_t)colBase * NH;

  v8f acc[9];
#pragma unroll
  for (int t = 0; t < 9; ++t) acc[t] = (v8f){0,0,0,0,0,0,0,0};

  b2_stage(bsrc, 0, tid, smem.bbuf[0]);          // prologue: stage k=0 tile
  wait_async_zero();
  __syncthreads();

  v16h a = make_frag(ap);
  const int lboff = (cg * 128 + m) * BSTR + koff;
#pragma unroll
  for (int kidx = 0; kidx < 15; ++kidx) {        // fully unrolled steady loop
    const int k = kidx * 32;
    const int cur = kidx & 1;
    b2_stage(bsrc, k + 32, tid, smem.bbuf[cur ^ 1]);  // async: issue only
    v16h an = make_frag(ap + k + 32);            // next A frag
    __builtin_amdgcn_sched_barrier(0);           // pin issue before the chain

    mma_chain<9>(&smem.bbuf[cur][lboff], a, acc);
    a = an;
    wait_async_zero();
    __syncthreads();
  }
  mma_chain<9>(&smem.bbuf[1][lboff], a, acc);    // peeled final k-step (buf 1)
  __syncthreads();                               // fence before union reuse

  const int sgBase = blockIdx.y * 16;

  // 2 phases of 32 rows: waves 4p..4p+3 dump their tiles, then all 256
  // threads run the spline epilogue on those 32 rows x 16 channels.
  for (int p = 0; p < 2; ++p) {
    if ((wave >> 2) == p) {
      int lr = ((wave >> 1) & 1) * 16 + (hw << 3);   // local row base (0..31)
#pragma unroll
      for (int t = 0; t < 9; ++t) {
        int c = (cg * 8 + t) * 16 + m;
        float bias = b2[colBase + c];
#pragma unroll
        for (int r = 0; r < 8; ++r)
          smem.ep.net_s[(lr + r) * 272 + c] = tanhf(acc[t][r] + bias);
      }
    }
    __syncthreads();

    int lrow = tid >> 3;                         // 0..31
    int s0   = (tid & 7) * 2;                    // 2 channels per thread
    int rowg = rowBase + p * 32 + lrow;
    float lsum = 0.f;
#pragma unroll
    for (int ss = 0; ss < 2; ++ss) {
      int s = s0 + ss;
      const float* nets = &smem.ep.net_s[lrow * 272 + s * 17];
      // softmax over w_raw (bounded in (-1,1): no max-shift needed)
      float w[8]; float sw = 0.f;
#pragma unroll
      for (int j = 0; j < 8; ++j) { w[j] = expf(nets[9 + j]); sw += w[j]; }
      float invsw = 1.0f / sw;
#pragma unroll
      for (int j = 0; j < 8; ++j) w[j] *= invsw;
      float eh[9];
#pragma unroll
      for (int i = 0; i < 9; ++i) eh[i] = expf(nets[i]);
      float denom = 0.f;
#pragma unroll
      for (int j = 0; j < 8; ++j) denom += 0.5f * w[j] * (eh[j] + eh[j+1]);
      float invd = 1.0f / denom;
      float h[9];
#pragma unroll
      for (int i = 0; i < 9; ++i) h[i] = eh[i] * invd;

      float xb = x_input[rowg * (2*NS) + NS + sgBase + s];

      // running cumsum + register-resident bin select (k = last j with
      // x_knots[j] < xb, clipped to [0,7]); avoids dynamic array indexing.
      float xk = -EPSV, pk = 0.f;
      float x_km1 = -EPSV, phi_km1 = 0.f;
      float w_k = w[0], h_k = h[0], h_k1 = h[1];
#pragma unroll
      for (int j = 0; j < 8; ++j) {
        if (xk < xb) { x_km1 = xk; phi_km1 = pk;
                       w_k = w[j]; h_k = h[j]; h_k1 = h[j+1]; }
        xk += w[j];
        pk += 0.5f * w[j] * (h[j] + h[j+1]);
      }
      float alpha = (xb - x_km1) / w_k;
      float phi = phi_km1 + alpha * h_k * w_k
                + 0.5f * alpha * alpha * (h_k1 - h_k) * w_k;
      out[rowg * (2*NS) + NS + sgBase + s] = phi;
      lsum += logf(h_k + alpha * (h_k1 - h_k));
    }
    smem.ep.red[tid] = lsum;
    __syncthreads();
    if ((tid & 7) == 0) {
      float r8 = 0.f;
#pragma unroll
      for (int q = 0; q < 8; ++q) r8 += smem.ep.red[tid + q];
      part[rowg * 32 + blockIdx.y] = r8;         // 16 channels' log sum
    }
    __syncthreads();
  }
}

// ---------------- finalize: new_log_density = ld - sum(partials) ------------
__global__ void finalize_logden_kernel(const float* __restrict__ log_density,
                                       const float* __restrict__ part,
                                       float* __restrict__ out_ld) {
  int b = blockIdx.x * 256 + threadIdx.x;        // 8192 exactly
  float s = 0.f;
#pragma unroll
  for (int j = 0; j < 32; ++j) s += part[b * 32 + j];
  out_ld[b] = log_density[b] - s;
}

extern "C" void kernel_launch(void* const* d_in, const int* in_sizes, int n_in,
                              void* d_out, int out_size, void* d_ws, size_t ws_size,
                              hipStream_t stream) {
  const float* x_input     = (const float*)d_in[0];
  const float* log_density = (const float*)d_in[1];
  const float* W1          = (const float*)d_in[2];
  const float* b1          = (const float*)d_in[3];
  const float* W2          = (const float*)d_in[4];
  const float* b2          = (const float*)d_in[5];
  float* out = (float*)d_out;

  char* ws = (char*)d_ws;
  _Float16* xah  = (_Float16*)(ws + WS_XAH);
  _Float16* hid  = (_Float16*)(ws + WS_HID);
  _Float16* w1t  = (_Float16*)(ws + WS_W1T);
  _Float16* w2t  = (_Float16*)(ws + WS_W2T);
  float*    part = (float*)(ws + WS_PART);

  prep_x_kernel<<<(NB * NS) / 256, 256, 0, stream>>>(x_input, xah, out);
  prep_wt_kernel<<<(NH * NS) / 256, 256, 0, stream>>>(W1, w1t, NH);
  prep_wt_kernel<<<(NCOL * NS) / 256, 256, 0, stream>>>(W2, w2t, NCOL);

  gemm1_tanh_kernel<<<dim3(NB / 128, NH / 128), 256, 0, stream>>>(xah, w1t, b1, hid);

  gemm2_spline_kernel<<<dim3(NB / 64, NCOL / 272), 256, 0, stream>>>(
      hid, w2t, b2, x_input, out, part);

  finalize_logden_kernel<<<NB / 256, 256, 0, stream>>>(
      log_density, part, out + (size_t)NB * (2 * NS));
}
